// PPDocLayoutV2ReadingOrderSelfAttention_1760936592079
// MI455X (gfx1250) — matrix operand order repaired and checked by reading, not compiled, not run on hardware
//
#include <hip/hip_runtime.h>

#define S_  1024
#define H_  512
#define NH_ 8
#define HD_ 64
#define B_  8

typedef __attribute__((ext_vector_type(16))) _Float16 v16h;
typedef __attribute__((ext_vector_type(8)))  _Float16 v8h;
typedef __attribute__((ext_vector_type(8)))  float    v8f;

__device__ __forceinline__ v16h cat16(v8h a, v8h b) {
  return __builtin_shufflevector(a, b, 0,1,2,3,4,5,6,7,8,9,10,11,12,13,14,15);
}

// 8 consecutive f32 -> 8 f16 (pointer must be 16B aligned)
__device__ __forceinline__ v8h cvt8(const float* __restrict__ p) {
  const float4* p4 = (const float4*)p;
  float4 x = p4[0], y = p4[1];
  v8h r;
  r[0]=(_Float16)x.x; r[1]=(_Float16)x.y; r[2]=(_Float16)x.z; r[3]=(_Float16)x.w;
  r[4]=(_Float16)y.x; r[5]=(_Float16)y.y; r[6]=(_Float16)y.z; r[7]=(_Float16)y.w;
  return r;
}

__device__ __forceinline__ v8f wmma_f16(v16h a, v16h b, v8f c) {
  // D = A(16x32 f16) * B(32x16 f16) + C(16x16 f32)
  return __builtin_amdgcn_wmma_f32_16x16x32_f16(false, a, false, b, (short)0, c,
                                                false, false);
}

__device__ __forceinline__ float rmax16(float v) {
  v = fmaxf(v, __shfl_xor(v, 1, 32));
  v = fmaxf(v, __shfl_xor(v, 2, 32));
  v = fmaxf(v, __shfl_xor(v, 4, 32));
  v = fmaxf(v, __shfl_xor(v, 8, 32));
  return v;
}
__device__ __forceinline__ float rsum16(float v) {
  v += __shfl_xor(v, 1, 32);
  v += __shfl_xor(v, 2, 32);
  v += __shfl_xor(v, 4, 32);
  v += __shfl_xor(v, 8, 32);
  return v;
}

// ---------------------------------------------------------------------------
// Kernel 1: QKV projection.  grid=(M/16, 2, 3), block=128 (4 waves).
// Each wave computes a 16x64 output tile with WMMA f16 (f32 accumulate).
// z=0 -> Q (pre-scaled by 1/sqrt(HD)), z=1 -> K, z=2 -> V stored transposed.
// ---------------------------------------------------------------------------
__global__ __launch_bounds__(128) void qkv_kernel(
    const float* __restrict__ X,
    const float* __restrict__ Wq, const float* __restrict__ bq,
    const float* __restrict__ Wk, const float* __restrict__ bk,
    const float* __restrict__ Wv, const float* __restrict__ bv,
    _Float16* __restrict__ qws, _Float16* __restrict__ kws,
    _Float16* __restrict__ vtws)
{
  const int lane  = threadIdx.x & 31;
  const int wave  = threadIdx.x >> 5;
  const int z     = blockIdx.z;
  const int m0    = blockIdx.x * 16;
  const int nb    = blockIdx.y * 256 + wave * 64;
  const int mrow  = lane & 15;
  const bool hi   = lane >= 16;
  const int koff  = hi ? 8 : 0;       // unused; A-operand also uses kb16
  const int kb16  = hi ? 16 : 0;      // B-operand kdim half

  const float* Wz = (z == 0) ? Wq : ((z == 1) ? Wk : Wv);
  const float* bz = (z == 0) ? bq : ((z == 1) ? bk : bv);

  const float* xrow = X + (size_t)(m0 + mrow) * H_;

  v8f c[4];
#pragma unroll
  for (int t = 0; t < 4; ++t) c[t] = (v8f){0,0,0,0,0,0,0,0};

#pragma unroll 4
  for (int k0 = 0; k0 < H_; k0 += 32) {
    // A: X tile 16x32 (lane m holds K {kb16..kb16+15})
    v16h a = cat16(cvt8(xrow + k0 + kb16), cvt8(xrow + k0 + kb16 + 8));
    const int kb = k0 + kb16;  // B: lane holds 16 contiguous kdim values
#pragma unroll
    for (int t = 0; t < 4; ++t) {
      const float* wrow = Wz + (size_t)(nb + t * 16 + mrow) * H_;
      v16h bm = cat16(cvt8(wrow + kb), cvt8(wrow + kb + 8));
      c[t] = wmma_f16(a, bm, c[t]);
    }
  }

  const int rowhi = hi ? 8 : 0;
#pragma unroll
  for (int t = 0; t < 4; ++t) {
    const int n    = nb + t * 16 + mrow;  // C layout: col = lane%16
    const float bi = bz[n];
    const int hh   = n >> 6;
    const int d    = n & 63;
#pragma unroll
    for (int r = 0; r < 8; ++r) {
      const int m  = m0 + r + rowhi;      // C layout: row = vgpr (+8 hi lanes)
      const int bb = m >> 10;
      const int s  = m & 1023;
      float v = c[t][r] + bi;
      if (z == 0) {
        qws[((size_t)(bb * NH_ + hh) * S_ + s) * HD_ + d] =
            (_Float16)(v * 0.125f);  // 1/sqrt(64)
      } else if (z == 1) {
        kws[((size_t)(bb * NH_ + hh) * S_ + s) * HD_ + d] = (_Float16)v;
      } else {
        vtws[((size_t)(bb * NH_ + hh) * HD_ + d) * S_ + s] = (_Float16)v;
      }
    }
  }
}

// ---------------------------------------------------------------------------
// Kernel 2: flash-style attention.  grid=(S/16, NH, B), block=32 (1 wave).
// Each wave owns 16 query rows x full HD=64, streams keys in chunks of 32,
// online softmax, rel_2d_pos + mask fused in f32.  Probs transposed through
// 1.25 KB of LDS (padded stride for aligned b128 loads).
// ---------------------------------------------------------------------------
__global__ __launch_bounds__(32) void attn_kernel(
    const _Float16* __restrict__ qws, const _Float16* __restrict__ kws,
    const _Float16* __restrict__ vtws, const float* __restrict__ rel,
    const float* __restrict__ amask, float* __restrict__ out)
{
  __shared__ _Float16 Plds[16 * 40];  // 16 rows x 32 cols, stride 40 (80B)

  const int lane  = threadIdx.x & 31;
  const int q0    = blockIdx.x * 16;
  const int h     = blockIdx.y;
  const int b     = blockIdx.z;
  const int bh    = b * NH_ + h;
  const int mrow  = lane & 15;
  const bool hi   = lane >= 16;
  const int koff  = hi ? 8 : 0;
  const int rowhi = hi ? 8 : 0;
  const int sel16 = hi ? 16 : 0;

  // Q tile as two A-operands (d 0..31 and d 32..63)
  const _Float16* qrow = qws + ((size_t)bh * S_ + q0 + mrow) * HD_;
  v16h aq0 = cat16(*(const v8h*)(qrow + sel16),      *(const v8h*)(qrow + sel16 + 8));
  v16h aq1 = cat16(*(const v8h*)(qrow + 32 + sel16), *(const v8h*)(qrow + 40 + sel16));

  float mi[8], li[8];
  v8f acc[4];
#pragma unroll
  for (int r = 0; r < 8; ++r) { mi[r] = -1e30f; li[r] = 0.f; }
#pragma unroll
  for (int t = 0; t < 4; ++t) acc[t] = (v8f){0,0,0,0,0,0,0,0};

  const float* relbase = rel + ((size_t)bh * S_ + q0) * S_;
  const float* maskrow = amask + (size_t)b * S_;

  for (int j0 = 0; j0 < S_; j0 += 32) {
    // ---- scores: two 16-key tiles, each = QK^T over d=64 (2 WMMAs) ----
    v8f c0 = (v8f){0,0,0,0,0,0,0,0};
    v8f c1 = (v8f){0,0,0,0,0,0,0,0};
    {
      const _Float16* krow = kws + ((size_t)bh * S_ + j0 + mrow) * HD_;
      c0 = wmma_f16(aq0, *(const v16h*)(krow + sel16), c0);
      c0 = wmma_f16(aq1, *(const v16h*)(krow + 32 + sel16), c0);
    }
    {
      const _Float16* krow = kws + ((size_t)bh * S_ + j0 + 16 + mrow) * HD_;
      c1 = wmma_f16(aq0, *(const v16h*)(krow + sel16), c1);
      c1 = wmma_f16(aq1, *(const v16h*)(krow + 32 + sel16), c1);
    }

    // ---- fuse rel bias + mask (the HBM-bound stream: 256 MB total) ----
    const float mk0 = maskrow[j0 + mrow];
    const float mk1 = maskrow[j0 + 16 + mrow];
    float s0[8], s1[8];
#pragma unroll
    for (int r = 0; r < 8; ++r) {
      const float* rr = relbase + (size_t)(r + rowhi) * S_ + j0 + mrow;
      s0[r] = c0[r] + rr[0]  + mk0;
      s1[r] = c1[r] + rr[16] + mk1;
    }

    // ---- online softmax (per row = per C-layout VGPR, 16-lane reduce) ----
#pragma unroll
    for (int r = 0; r < 8; ++r) {
      float tmax = rmax16(fmaxf(s0[r], s1[r]));
      float mn   = fmaxf(mi[r], tmax);
      float corr = __expf(mi[r] - mn);
      mi[r] = mn;
      float p0 = __expf(s0[r] - mn);
      float p1 = __expf(s1[r] - mn);
      li[r] = li[r] * corr + rsum16(p0 + p1);
#pragma unroll
      for (int t = 0; t < 4; ++t) acc[t][r] *= corr;
      Plds[(r + rowhi) * 40 + mrow]      = (_Float16)p0;
      Plds[(r + rowhi) * 40 + 16 + mrow] = (_Float16)p1;
    }
    __syncthreads();  // single-wave WG -> S_NOP; LDS ops are in-order per wave

    // ---- P (16x32) as A-operand, V^T rows as B-operand, 4 PV WMMAs ----
    const _Float16* prow = &Plds[mrow * 40];
    v16h ap = cat16(*(const v8h*)(prow + sel16), *(const v8h*)(prow + 8 + sel16));
#pragma unroll
    for (int t = 0; t < 4; ++t) {
      const _Float16* vrow = vtws + ((size_t)bh * HD_ + t * 16 + mrow) * S_;
      v16h bv = *(const v16h*)(vrow + j0 + sel16);
      acc[t] = wmma_f16(ap, bv, acc[t]);
    }
    __syncthreads();
  }

  // ---- normalize and write ctx [B,S,H] ----
#pragma unroll
  for (int t = 0; t < 4; ++t) {
#pragma unroll
    for (int r = 0; r < 8; ++r) {
      const float o  = acc[t][r] / li[r];
      const int srow = q0 + r + rowhi;
      out[((size_t)b * S_ + srow) * H_ + h * HD_ + t * 16 + mrow] = o;
    }
  }
}

extern "C" void kernel_launch(void* const* d_in, const int* in_sizes, int n_in,
                              void* d_out, int out_size, void* d_ws, size_t ws_size,
                              hipStream_t stream) {
  (void)in_sizes; (void)n_in; (void)out_size; (void)ws_size;
  const float* X     = (const float*)d_in[0];
  const float* amask = (const float*)d_in[1];
  const float* rel   = (const float*)d_in[2];
  const float* Wq    = (const float*)d_in[3];
  const float* bq    = (const float*)d_in[4];
  const float* Wk    = (const float*)d_in[5];
  const float* bk    = (const float*)d_in[6];
  const float* Wv    = (const float*)d_in[7];
  const float* bv    = (const float*)d_in[8];
  float* out = (float*)d_out;

  const size_t per = (size_t)B_ * NH_ * S_ * HD_;  // 4M elems (8 MB f16) each
  _Float16* qws  = (_Float16*)d_ws;
  _Float16* kws  = qws + per;
  _Float16* vtws = kws + per;

  qkv_kernel<<<dim3((B_ * S_) / 16, 2, 3), 128, 0, stream>>>(
      X, Wq, bq, Wk, bk, Wv, bv, qws, kws, vtws);
  attn_kernel<<<dim3(S_ / 16, NH_, B_), 32, 0, stream>>>(
      qws, kws, vtws, rel, amask, out);
}
